// GQA_68221260530046
// MI455X (gfx1250) — compile-verified, matrix operand order checked by reading, not run on hardware
//
#include <hip/hip_runtime.h>
#include <hip/hip_bf16.h>

// ---------------------------------------------------------------------------
// GQA attention layer for MI455X (gfx1250, wave32, WMMA).
// Matmuls: v_wmma_f32_16x16x32_f16. GEMM staging: double-buffered
// global_load_async_to_lds_b128 (ASYNCcnt, in-order completion) so global->LDS
// copies for tile t+1 overlap the WMMAs of tile t.
// Fragments: b128 pairs (16-bit K-striping = two contiguous 8-half runs).
// ---------------------------------------------------------------------------

typedef __attribute__((ext_vector_type(16))) _Float16 v16h;
typedef __attribute__((ext_vector_type(8)))  _Float16 v8h;
typedef __attribute__((ext_vector_type(8)))  float    v8f;

#define S_LEN 2048
#define DMODEL 2048
#define NQH 32
#define NKVH 8
#define HEADD 64

__device__ __forceinline__ v16h combine16(v8h lo, v8h hi) {
    return __builtin_shufflevector(lo, hi, 0, 1, 2, 3, 4, 5, 6, 7,
                                           8, 9, 10, 11, 12, 13, 14, 15);
}

// Fragment load from a row-contiguous f16 tile: `p` points at this lane's row.
// lanes 0-15 (hi=0): K = {kc+0..7, kc+16..23}; lanes 16-31: +8.
__device__ __forceinline__ v16h load_frag(const _Float16* p, int kc, int hi) {
    v8h lo = *(const v8h*)(p + kc + hi * 8);
    v8h hh = *(const v8h*)(p + kc + 16 + hi * 8);
    return combine16(lo, hh);
}

#define WMMA_F16(A, B, C) \
    __builtin_amdgcn_wmma_f32_16x16x32_f16(false, (A), false, (B), (short)0, (C), false, false)

// Async global -> LDS copy, 16B per lane, tracked by ASYNCcnt (CDNA5 §15.18).
__device__ __forceinline__ void async_copy_b128(void* lds_addr, const void* gaddr) {
    unsigned lds_off = (unsigned)(uintptr_t)lds_addr;  // low 32 bits = LDS offset
    asm volatile("global_load_async_to_lds_b128 %0, %1, off"
                 :: "v"(lds_off), "v"(gaddr) : "memory");
}
__device__ __forceinline__ void wait_async_le(int n) {  // n must be literal-ish
    if (n == 0) asm volatile("s_wait_asynccnt 0x0" ::: "memory");
    else        asm volatile("s_wait_asynccnt 0x8" ::: "memory");
}
// Compiler-only ordering fence (hardware keeps same-wave DS ops in order).
__device__ __forceinline__ void wave_lds_fence() {
    asm volatile("" ::: "memory");
}

// ---------------------------------------------------------------------------
// f32 -> f16 convert (row-major preserved).
// ---------------------------------------------------------------------------
__global__ __launch_bounds__(256) void cvt_f16(const float* __restrict__ src,
                                               _Float16* __restrict__ dst) {
    size_t i = ((size_t)blockIdx.x * 256 + threadIdx.x) * 8;
    float4 a = *(const float4*)(src + i);
    float4 b = *(const float4*)(src + i + 4);
    v8h h;
    h[0] = (_Float16)a.x; h[1] = (_Float16)a.y; h[2] = (_Float16)a.z; h[3] = (_Float16)a.w;
    h[4] = (_Float16)b.x; h[5] = (_Float16)b.y; h[6] = (_Float16)b.z; h[7] = (_Float16)b.w;
    *(v8h*)(dst + i) = h;
}

// ---------------------------------------------------------------------------
// Transpose + convert: W[K][N] f32 -> Wt[N][K] f16. 32x32 tiles via LDS.
// ---------------------------------------------------------------------------
__global__ __launch_bounds__(256) void transpose_cvt(const float* __restrict__ W,
                                                     _Float16* __restrict__ Wt,
                                                     int K, int N) {
    __shared__ float tile[32][33];
    const int n0 = blockIdx.x * 32, k0 = blockIdx.y * 32;
    const int tx = threadIdx.x, ty = threadIdx.y;
#pragma unroll
    for (int i = 0; i < 4; ++i)
        tile[ty + i * 8][tx] = W[(size_t)(k0 + ty + i * 8) * N + n0 + tx];
    __syncthreads();
#pragma unroll
    for (int i = 0; i < 4; ++i)
        Wt[(size_t)(n0 + ty + i * 8) * K + k0 + tx] = (_Float16)tile[tx][ty + i * 8];
}

// ---------------------------------------------------------------------------
// GEMM: C[MxN] f32 = A[MxK] f16 (row-major) * Bt[NxK] f16 (pre-transposed).
// 128 threads (4 waves), 64x64 tile, K-step 64. Double-buffered async LDS
// staging: copies for the next K-tile overlap the current tile's 8 WMMAs.
// ---------------------------------------------------------------------------
#define TS 72  // LDS row stride in halfs (144B, 16B-aligned, padded)
__global__ __launch_bounds__(128) void gemm_wmma_f16(
    const _Float16* __restrict__ A, const _Float16* __restrict__ Bt,
    float* __restrict__ C, int M, int N, int K) {
    __shared__ __align__(16) _Float16 As[2][64 * TS];
    __shared__ __align__(16) _Float16 Bs[2][64 * TS];

    const int tid  = threadIdx.x;
    const int lane = tid & 31;
    const int wave = tid >> 5;
    const int r16  = lane & 15, hi = lane >> 4;
    const int row0 = blockIdx.y * 64;
    const int col0 = blockIdx.x * 64;

    // Issue the 8 async copies (4 A + 4 B chunks of 8 halfs) for one K-tile.
    auto issue_tile = [&](int buf, int k0) {
#pragma unroll
        for (int t = 0; t < 4; ++t) {
            int idx = tid + t * 128;
            int r = idx >> 3, c = (idx & 7) * 8;
            async_copy_b128(As[buf] + r * TS + c,
                            A + (size_t)(row0 + r) * K + k0 + c);
            async_copy_b128(Bs[buf] + r * TS + c,
                            Bt + (size_t)(col0 + r) * K + k0 + c);
        }
    };

    v8f acc[4] = {};

    issue_tile(0, 0);  // prologue
    for (int k0 = 0; k0 < K; k0 += 64) {
        const int cur = (k0 >> 6) & 1;
        if (k0 + 64 < K) {
            issue_tile(cur ^ 1, k0 + 64);  // overlap with this tile's compute
            wait_async_le(8);              // drain the 8 older copies (cur tile)
        } else {
            wait_async_le(0);
        }
        __syncthreads();

        // Preload every fragment of this K-step, then run the 8 WMMAs.
        v16h afr[2], bfr[2][4];
#pragma unroll
        for (int kk = 0; kk < 2; ++kk) {
            afr[kk] = load_frag(As[cur] + (wave * 16 + r16) * TS, kk * 32, hi);
#pragma unroll
            for (int c = 0; c < 4; ++c)
                bfr[kk][c] = load_frag(Bs[cur] + (c * 16 + r16) * TS, kk * 32, hi);
        }
#pragma unroll
        for (int kk = 0; kk < 2; ++kk)
#pragma unroll
            for (int c = 0; c < 4; ++c)
                acc[c] = WMMA_F16(afr[kk], bfr[kk][c], acc[c]);
        __syncthreads();
    }

    // C/D layout: VGPR r, lane -> row = r + 8*hi, col = lane&15.
#pragma unroll
    for (int c = 0; c < 4; ++c)
#pragma unroll
        for (int r = 0; r < 8; ++r) {
            int row = row0 + wave * 16 + r + 8 * hi;
            int col = col0 + c * 16 + r16;
            C[(size_t)row * N + col] = acc[c][r];
        }
}

// ---------------------------------------------------------------------------
// Fused RMSNorm + RoPE + f32->f16 repack. One wave per (s, head) row; lane
// owns the RoPE pair (d, d+32). mode 0/1: norm+rope -> [head][S][64].
// mode 2: plain convert -> [head][64][S] (transposed, for PV B-fragments).
// ---------------------------------------------------------------------------
__global__ __launch_bounds__(128) void rope_cvt(
    const float* __restrict__ src, const float* __restrict__ cosd,
    const float* __restrict__ sind, const float* __restrict__ g,
    _Float16* __restrict__ dst, int nheads, int mode) {
    const int lane = threadIdx.x & 31;
    const int wave = threadIdx.x >> 5;
    const int gid  = blockIdx.x * 4 + wave;
    const int s    = gid / nheads;
    const int head = gid - s * nheads;

    const float* row = src + (size_t)s * (nheads * HEADD) + head * HEADD;
    float x1 = row[lane];
    float x2 = row[lane + 32];

    if (mode != 2) {
        float ss = x1 * x1 + x2 * x2;
#pragma unroll
        for (int off = 16; off > 0; off >>= 1) ss += __shfl_xor(ss, off, 32);
        float inv = rsqrtf(ss * (1.0f / 64.0f) + 1e-6f);
        x1 = x1 * inv * g[lane];
        x2 = x2 * inv * g[lane + 32];
        float cv = cosd[s * 32 + lane];
        float sv = sind[s * 32 + lane];
        float o1 = x1 * cv - x2 * sv;
        float o2 = x1 * sv + x2 * cv;
        x1 = o1; x2 = o2;
        _Float16* drow = dst + ((size_t)head * S_LEN + s) * HEADD;
        drow[lane]      = (_Float16)x1;
        drow[lane + 32] = (_Float16)x2;
    } else {
        _Float16* dbase = dst + (size_t)head * HEADD * S_LEN;
        dbase[(size_t)lane * S_LEN + s]        = (_Float16)x1;
        dbase[(size_t)(lane + 32) * S_LEN + s] = (_Float16)x2;
    }
}

// ---------------------------------------------------------------------------
// Flash attention: grid (S/64, NQH), 128 threads (4 waves), 16 q-rows/wave.
// The P staging buffer is strictly per-wave, and a wave's DS instructions
// execute in order (ISA 7.3), so no workgroup barrier is needed in the KV
// loop -- only a compiler fence. Waves run fully decoupled.
// ---------------------------------------------------------------------------
__global__ __launch_bounds__(128) void attn_flash(
    const _Float16* __restrict__ Qh, const _Float16* __restrict__ Kh,
    const _Float16* __restrict__ Vt, _Float16* __restrict__ Oh) {
    __shared__ __align__(16) _Float16 P[4][16 * 32];

    const int lane = threadIdx.x & 31;
    const int wave = threadIdx.x >> 5;
    const int r16  = lane & 15, hi = lane >> 4;
    const int qblk = blockIdx.x;
    const int h    = blockIdx.y;
    const int kvh  = h >> 2;  // NREP = 4

    const _Float16* Qb = Qh + (size_t)h   * S_LEN * HEADD;
    const _Float16* Kb = Kh + (size_t)kvh * S_LEN * HEADD;
    const _Float16* Vb = Vt + (size_t)kvh * HEADD * S_LEN;

    const int qr0 = qblk * 64 + wave * 16;

    // Q fragments (HD chunks), pre-scaled by 1/sqrt(64)=0.125.
    const _Float16* qrow = Qb + (size_t)(qr0 + r16) * HEADD;
    v16h aQ0 = load_frag(qrow, 0, hi);
    v16h aQ1 = load_frag(qrow, 32, hi);
#pragma unroll
    for (int j = 0; j < 16; ++j) {
        aQ0[j] = (_Float16)((float)aQ0[j] * 0.125f);
        aQ1[j] = (_Float16)((float)aQ1[j] * 0.125f);
    }

    float m[8], l[8];
#pragma unroll
    for (int r = 0; r < 8; ++r) { m[r] = -1.0e30f; l[r] = 0.0f; }
    v8f acc[4] = {};

    const int kv_end = qblk * 64 + 64;  // causal coverage for this block
    for (int t0 = 0; t0 < kv_end; t0 += 32) {
        // scores: preload 4 K-fragments, then 4 WMMAs.
        const _Float16* kr0 = Kb + (size_t)(t0 + r16) * HEADD;
        const _Float16* kr1 = Kb + (size_t)(t0 + 16 + r16) * HEADD;
        v16h bk[4];
        bk[0] = load_frag(kr0, 0, hi);
        bk[1] = load_frag(kr0, 32, hi);
        bk[2] = load_frag(kr1, 0, hi);
        bk[3] = load_frag(kr1, 32, hi);
        v8f c0 = {}, c1 = {};
        c0 = WMMA_F16(aQ0, bk[0], c0);
        c0 = WMMA_F16(aQ1, bk[1], c0);
        c1 = WMMA_F16(aQ0, bk[2], c1);
        c1 = WMMA_F16(aQ1, bk[3], c1);

        // Causal mask + online softmax (16-lane row reductions).
#pragma unroll
        for (int r = 0; r < 8; ++r) {
            int grow = qr0 + r + 8 * hi;
            float s0 = (t0 + r16      <= grow) ? c0[r] : -1.0e30f;
            float s1 = (t0 + 16 + r16 <= grow) ? c1[r] : -1.0e30f;
            float mx = fmaxf(s0, s1);
#pragma unroll
            for (int off = 1; off < 16; off <<= 1) mx = fmaxf(mx, __shfl_xor(mx, off, 32));
            float mn  = fmaxf(m[r], mx);
            float fac = __expf(m[r] - mn);
            float p0  = __expf(s0 - mn);
            float p1  = __expf(s1 - mn);
            float ps  = p0 + p1;
#pragma unroll
            for (int off = 1; off < 16; off <<= 1) ps += __shfl_xor(ps, off, 32);
            l[r] = l[r] * fac + ps;
            m[r] = mn;
#pragma unroll
            for (int c = 0; c < 4; ++c) acc[c][r] = acc[c][r] * fac;
            P[wave][(r + 8 * hi) * 32 + r16]      = (_Float16)p0;
            P[wave][(r + 8 * hi) * 32 + 16 + r16] = (_Float16)p1;
        }
        wave_lds_fence();  // per-wave buffer; HW keeps wave DS ops in order

        // P*V: preload P A-fragment + 4 V B-fragments, then 4 WMMAs.
        v16h aP = load_frag(&P[wave][r16 * 32], 0, hi);
        v16h bv[4];
#pragma unroll
        for (int c = 0; c < 4; ++c)
            bv[c] = load_frag(Vb + (size_t)(c * 16 + r16) * S_LEN + t0, 0, hi);
#pragma unroll
        for (int c = 0; c < 4; ++c)
            acc[c] = WMMA_F16(aP, bv[c], acc[c]);
        wave_lds_fence();  // keep next iteration's P stores after the aP load
    }

#pragma unroll
    for (int r = 0; r < 8; ++r) {
        float inv = 1.0f / l[r];
        int grow  = qr0 + r + 8 * hi;
        _Float16* orow = Oh + (size_t)grow * DMODEL + h * HEADD;
#pragma unroll
        for (int c = 0; c < 4; ++c) orow[c * 16 + r16] = (_Float16)(acc[c][r] * inv);
    }
}

// ---------------------------------------------------------------------------
extern "C" void kernel_launch(void* const* d_in, const int* in_sizes, int n_in,
                              void* d_out, int out_size, void* d_ws, size_t ws_size,
                              hipStream_t stream) {
    const float* x    = (const float*)d_in[0];
    const float* cosd = (const float*)d_in[1];
    const float* sind = (const float*)d_in[2];
    const float* g_q  = (const float*)d_in[3];
    const float* g_k  = (const float*)d_in[4];
    const float* Wq   = (const float*)d_in[5];
    const float* Wk   = (const float*)d_in[6];
    const float* Wv   = (const float*)d_in[7];
    const float* Wo   = (const float*)d_in[8];
    float* out = (float*)d_out;

    char* ws = (char*)d_ws;
    const size_t MB = 1048576;
    _Float16* xh  = (_Float16*)(ws);             // 2048x2048 f16 (8MB)
    _Float16* Wqt = (_Float16*)(ws + 8  * MB);   // [2048][2048] f16 (8MB)
    _Float16* Wkt = (_Float16*)(ws + 16 * MB);   // [512][2048]  f16 (2MB)
    _Float16* Wvt = (_Float16*)(ws + 18 * MB);   // [512][2048]  f16 (2MB)
    _Float16* Wot = (_Float16*)(ws + 20 * MB);   // [2048][2048] f16 (8MB)
    float*    Qf  = (float*)   (ws + 28 * MB);   // 2048x2048 f32 (16MB)
    float*    Kf  = (float*)   (ws + 44 * MB);   // 2048x512  f32 (4MB)
    float*    Vf  = (float*)   (ws + 48 * MB);   // 2048x512  f32 (4MB)
    _Float16* Qhh = (_Float16*)(ws + 52 * MB);   // [32][2048][64] f16 (8MB)
    _Float16* Khh = (_Float16*)(ws + 60 * MB);   // [8][2048][64]  f16 (2MB)
    _Float16* Vth = (_Float16*)(ws + 62 * MB);   // [8][64][2048]  f16 (2MB)
    _Float16* Oh  = (_Float16*)(ws + 64 * MB);   // 2048x2048 f16 (8MB)

    dim3 b128(128), b256(256), bT(32, 8);

    // One-time precision/layout prep (bandwidth-trivial vs the GEMMs).
    cvt_f16<<<(S_LEN * DMODEL) / (256 * 8), b256, 0, stream>>>(x, xh);
    transpose_cvt<<<dim3(DMODEL / 32, DMODEL / 32), bT, 0, stream>>>(Wq, Wqt, DMODEL, DMODEL);
    transpose_cvt<<<dim3(512 / 32,    DMODEL / 32), bT, 0, stream>>>(Wk, Wkt, DMODEL, 512);
    transpose_cvt<<<dim3(512 / 32,    DMODEL / 32), bT, 0, stream>>>(Wv, Wvt, DMODEL, 512);
    transpose_cvt<<<dim3(DMODEL / 32, DMODEL / 32), bT, 0, stream>>>(Wo, Wot, DMODEL, DMODEL);

    // QKV projections (f16 WMMA, f32 out for the norm).
    gemm_wmma_f16<<<dim3(DMODEL / 64, S_LEN / 64), b128, 0, stream>>>(xh, Wqt, Qf, S_LEN, DMODEL, DMODEL);
    gemm_wmma_f16<<<dim3(512 / 64,    S_LEN / 64), b128, 0, stream>>>(xh, Wkt, Kf, S_LEN, 512, DMODEL);
    gemm_wmma_f16<<<dim3(512 / 64,    S_LEN / 64), b128, 0, stream>>>(xh, Wvt, Vf, S_LEN, 512, DMODEL);

    // RMSNorm + RoPE + repack to fragment-friendly f16 layouts.
    rope_cvt<<<(S_LEN * NQH)  / 4, b128, 0, stream>>>(Qf, cosd, sind, g_q, Qhh, NQH, 0);
    rope_cvt<<<(S_LEN * NKVH) / 4, b128, 0, stream>>>(Kf, cosd, sind, g_k, Khh, NKVH, 1);
    rope_cvt<<<(S_LEN * NKVH) / 4, b128, 0, stream>>>(Vf, cosd, sind, g_q, Vth, NKVH, 2);

    // Causal flash attention (f16 out).
    attn_flash<<<dim3(S_LEN / 64, NQH), b128, 0, stream>>>(Qhh, Khh, Vth, Oh);

    // Output projection.
    gemm_wmma_f16<<<dim3(DMODEL / 64, S_LEN / 64), b128, 0, stream>>>(Oh, Wot, out, S_LEN, DMODEL, DMODEL);
}
